// DGCNN_1254130450623
// MI455X (gfx1250) — matrix operand order repaired and checked by reading, not compiled, not run on hardware
//
#include <hip/hip_runtime.h>
#include <math.h>

#define NN    200000
#define NPG   400
#define GG    500
#define EE    3200000
#define HH    64
#define KK    60
#define DTOT  193      // H*L + 1
#define DENSE 832      // 26*32
#define POOLK 30       // K/2

typedef __attribute__((ext_vector_type(16))) _Float16 v16h;
typedef __attribute__((ext_vector_type(8)))  float    v8f;

__device__ __forceinline__ void atomAddF(float* p, float v) {
    __hip_atomic_fetch_add(p, v, __ATOMIC_RELAXED, __HIP_MEMORY_SCOPE_AGENT);
}

// ---------------- degree / norm ----------------
__global__ void k_init_deg(float* deg) {
    int i = blockIdx.x * blockDim.x + threadIdx.x;
    if (i < NN) deg[i] = 1.0f;                       // self-loop contributes 1
}
__global__ __launch_bounds__(256) void k_count_deg(const int* __restrict__ dst, float* deg) {
    int e = blockIdx.x * blockDim.x + threadIdx.x;
    if (e < EE) atomAddF(&deg[dst[e]], 1.0f);
}
__global__ void k_rsqrt_deg(float* dinv) {
    int i = blockIdx.x * blockDim.x + threadIdx.x;
    if (i < NN) dinv[i] = rsqrtf(dinv[i]);
}
__global__ __launch_bounds__(256) void k_edge_norm(const int* __restrict__ src, const int* __restrict__ dst,
                                                   const float* __restrict__ dinv, float* __restrict__ nrm) {
    int e = blockIdx.x * blockDim.x + threadIdx.x;
    if (e < EE) nrm[e] = dinv[src[e]] * dinv[dst[e]];
}

// ---------------- embedding ----------------
__global__ __launch_bounds__(256) void k_embed(const int* __restrict__ z, const float* __restrict__ tab,
                                               float* __restrict__ x0) {
    int t = blockIdx.x * blockDim.x + threadIdx.x;   // N*64 threads
    if (t >= NN * HH) return;
    int node = t >> 6, c = t & 63;
    x0[t] = tab[(size_t)z[node] * HH + c];
}

// ---------------- WMMA GEMM: Y[N,64] = X[N,64] @ W[64,64] ----------------
// 8 waves/block; wave w -> row-tile blockIdx.x*2 + (w>>2), col-tile (w&3)
__global__ __launch_bounds__(256) void k_gemm64_wmma(const float* __restrict__ X,
                                                     const float* __restrict__ W,
                                                     float* __restrict__ Y, int nrows) {
    const int wave = threadIdx.x >> 5;
    const int lane = threadIdx.x & 31;
    const int m0 = (blockIdx.x * 2 + (wave >> 2)) * 16;
    if (m0 >= nrows) return;                          // wave-uniform
    const int n0 = (wave & 3) * 16;
    const int g  = lane >> 4;                         // half-wave
    const int mr = lane & 15;
    v8f acc = {};
#pragma unroll
    for (int kk = 0; kk < 2; ++kk) {
        const int kb = kk * 32;
        v16h a, b;
        // A (16x32 f16): lane L (m=L&15,g=L>>4): elem i -> K=kb+16g+i
        const float* xr = X + (size_t)(m0 + mr) * HH + kb + 16 * g;
#pragma unroll
        for (int i = 0; i < 16; ++i) a[i] = (_Float16)xr[i];
        // B (32x16 f16): lane L (n=L&15,g=L>>4): elem i -> K=kb+16g+i
        const float* wr = W + (size_t)(kb + 16 * g) * HH + n0 + mr;
#pragma unroll
        for (int i = 0; i < 16; ++i) b[i] = (_Float16)wr[(size_t)i * HH];
        acc = __builtin_amdgcn_wmma_f32_16x16x32_f16(false, a, false, b, (short)0, acc, false, false);
    }
    // C/D: VGPR r -> M = r + 8*g, N = lane&15
    float* yr = Y + (size_t)(m0 + 8 * g) * HH + n0 + mr;
#pragma unroll
    for (int r = 0; r < 8; ++r) yr[(size_t)r * HH] = acc[r];
}

// ---------------- GCN scatter (H=64) ----------------
__global__ __launch_bounds__(256) void k_self64(const float* __restrict__ h, const float* __restrict__ dinv,
                                                float* __restrict__ out) {
    int t = blockIdx.x * blockDim.x + threadIdx.x;    // N*64
    if (t >= NN * HH) return;
    float di = dinv[t >> 6];
    out[t] = di * di * h[t];
}
__global__ __launch_bounds__(256) void k_scatter64(const int* __restrict__ src, const int* __restrict__ dst,
                                                   const float* __restrict__ nrm, const float* __restrict__ h,
                                                   float* __restrict__ out) {
    long long t = (long long)blockIdx.x * blockDim.x + threadIdx.x;  // E*16 threads
    int e = (int)(t >> 4);
    if (e >= EE) return;
    int c4 = (int)(t & 15);
    int s = src[e], d = dst[e];
    float w = nrm[e];
    const float4 hv = *(const float4*)(h + (size_t)s * HH + c4 * 4);
    float* o = out + (size_t)d * HH + c4 * 4;
    atomAddF(o + 0, w * hv.x);
    atomAddF(o + 1, w * hv.y);
    atomAddF(o + 2, w * hv.z);
    atomAddF(o + 3, w * hv.w);
}
__global__ __launch_bounds__(256) void k_tanh_bias64(float* __restrict__ x, const float* __restrict__ b) {
    int t = blockIdx.x * blockDim.x + threadIdx.x;    // N*64
    if (t < NN * HH) x[t] = tanhf(x[t] + b[t & 63]);
}

// ---------------- layer 4 (H=1) ----------------
__global__ void k_gemv64(const float* __restrict__ x, const float* __restrict__ w, float* __restrict__ y) {
    int i = blockIdx.x * blockDim.x + threadIdx.x;
    if (i >= NN) return;
    const float* r = x + (size_t)i * HH;
    float s = 0.f;
#pragma unroll
    for (int c = 0; c < HH; ++c) s += r[c] * w[c];
    y[i] = s;
}
__global__ void k_self1(const float* __restrict__ h, const float* __restrict__ dinv, float* __restrict__ out) {
    int i = blockIdx.x * blockDim.x + threadIdx.x;
    if (i < NN) { float di = dinv[i]; out[i] = di * di * h[i]; }
}
__global__ __launch_bounds__(256) void k_scatter1(const int* __restrict__ src, const int* __restrict__ dst,
                                                  const float* __restrict__ nrm, const float* __restrict__ h,
                                                  float* __restrict__ out) {
    int e = blockIdx.x * blockDim.x + threadIdx.x;
    if (e < EE) atomAddF(&out[dst[e]], nrm[e] * h[src[e]]);
}
__global__ void k_tanh_bias1(float* __restrict__ x, const float* __restrict__ b) {
    int i = blockIdx.x * blockDim.x + threadIdx.x;
    if (i < NN) x[i] = tanhf(x[i] + b[0]);
}

// ---------------- sort-pool: per-graph top-K by x4, stable desc ----------------
__global__ __launch_bounds__(256) void k_sortpool(const float* __restrict__ key, int* __restrict__ sel) {
    __shared__ float sk[NPG];
    int g = blockIdx.x;
    for (int j = threadIdx.x; j < NPG; j += 256) sk[j] = key[(size_t)g * NPG + j];
    __syncthreads();
    for (int j = threadIdx.x; j < NPG; j += 256) {
        float kj = sk[j];
        int rank = 0;
        for (int i = 0; i < NPG; ++i) {
            float ki = sk[i];
            rank += (ki > kj) || (ki == kj && i < j);   // stable argsort(-key)
        }
        if (rank < KK) sel[g * KK + rank] = g * NPG + j;
    }
}
__global__ __launch_bounds__(224) void k_gather(const int* __restrict__ sel,
                                                const float* __restrict__ x1, const float* __restrict__ x2,
                                                const float* __restrict__ x3, const float* __restrict__ x4,
                                                float* __restrict__ xp) {
    int row = blockIdx.x;                 // G*K rows
    int c = threadIdx.x;
    if (c >= DTOT) return;
    int node = sel[row];
    float v;
    if (c < 64)       v = x1[(size_t)node * HH + c];
    else if (c < 128) v = x2[(size_t)node * HH + c - 64];
    else if (c < 192) v = x3[(size_t)node * HH + c - 128];
    else              v = x4[node];
    xp[(size_t)row * DTOT + c] = v;
}

// ---------------- conv1 (per-slot 193-proj) + relu + maxpool2 ----------------
__global__ __launch_bounds__(256) void k_conv1pool(const float* __restrict__ xp, const float* __restrict__ w,
                                                   const float* __restrict__ b, float* __restrict__ out) {
    int t = blockIdx.x * blockDim.x + threadIdx.x;    // (g*16 + c)*30 + p
    if (t >= GG * 16 * POOLK) return;
    int p = t % POOLK; int gc = t / POOLK; int c = gc % 16; int g = gc / 16;
    const float* wr = w + c * DTOT;
    float m = 0.f;
#pragma unroll
    for (int q = 0; q < 2; ++q) {
        const float* row = xp + ((size_t)g * KK + 2 * p + q) * DTOT;
        float s = b[c];
        for (int d = 0; d < DTOT; ++d) s += row[d] * wr[d];
        s = fmaxf(s, 0.f);
        m = (q == 0) ? s : fmaxf(m, s);
    }
    out[t] = m;                                       // [G,16,30]
}

// ---------------- conv2: Conv1d(16,32,5) + relu ----------------
__global__ __launch_bounds__(256) void k_conv2(const float* __restrict__ x, const float* __restrict__ w,
                                               const float* __restrict__ b, float* __restrict__ out) {
    int t = blockIdx.x * blockDim.x + threadIdx.x;    // (g*32 + o)*26 + u
    if (t >= GG * 32 * 26) return;
    int u = t % 26; int go = t / 26; int o = go % 32; int g = go / 32;
    float s = b[o];
#pragma unroll
    for (int i = 0; i < 16; ++i) {
        const float* xr = x + (size_t)g * 480 + i * POOLK + u;
        const float* wr = w + (o * 16 + i) * 5;
#pragma unroll
        for (int r = 0; r < 5; ++r) s += xr[r] * wr[r];
    }
    out[t] = fmaxf(s, 0.f);                           // [G,32,26] == [G,832]
}

// ---------------- MLP head ----------------
__global__ __launch_bounds__(128) void k_lin1(const float* __restrict__ x, const float* __restrict__ w,
                                              const float* __restrict__ b, float* __restrict__ out) {
    int t = blockIdx.x * blockDim.x + threadIdx.x;    // g*128 + j
    if (t >= GG * 128) return;
    int j = t & 127; int g = t >> 7;
    const float* xr = x + (size_t)g * DENSE;
    float s = b[j];
    for (int d = 0; d < DENSE; ++d) s += xr[d] * w[(size_t)d * 128 + j];
    out[t] = fmaxf(s, 0.f);
}
__global__ void k_lin2(const float* __restrict__ x, const float* __restrict__ w,
                       const float* __restrict__ b, float* __restrict__ out) {
    int t = blockIdx.x * blockDim.x + threadIdx.x;    // g*2 + o
    if (t >= GG * 2) return;
    int o = t & 1; int g = t >> 1;
    const float* xr = x + (size_t)g * 128;
    float s = b[o];
#pragma unroll 8
    for (int d = 0; d < 128; ++d) s += xr[d] * w[d * 2 + o];
    out[t] = s;
}

static inline size_t alignup(size_t x) { return (x + 255) & ~(size_t)255; }

extern "C" void kernel_launch(void* const* d_in, const int* in_sizes, int n_in,
                              void* d_out, int out_size, void* d_ws, size_t ws_size,
                              hipStream_t stream) {
    (void)in_sizes; (void)n_in; (void)out_size; (void)ws_size;
    const int*   z        = (const int*)d_in[1];
    const int*   ei       = (const int*)d_in[2];      // [2,E]
    const float* z_table  = (const float*)d_in[4];
    const float* W0 = (const float*)d_in[5],  *b0 = (const float*)d_in[6];
    const float* W1 = (const float*)d_in[7],  *b1 = (const float*)d_in[8];
    const float* W2 = (const float*)d_in[9],  *b2 = (const float*)d_in[10];
    const float* W3 = (const float*)d_in[11], *b3 = (const float*)d_in[12];
    const float* c1w = (const float*)d_in[13], *c1b = (const float*)d_in[14];
    const float* c2w = (const float*)d_in[15], *c2b = (const float*)d_in[16];
    const float* l1w = (const float*)d_in[17], *l1b = (const float*)d_in[18];
    const float* l2w = (const float*)d_in[19], *l2b = (const float*)d_in[20];
    float* out = (float*)d_out;

    const int* esrc = ei;
    const int* edst = ei + EE;

    char* ws = (char*)d_ws;
    size_t off = 0;
    auto alloc = [&](size_t bytes) -> void* { void* p = ws + off; off = alignup(off + bytes); return p; };
    float* dinv = (float*)alloc((size_t)NN * 4);
    float* nrm  = (float*)alloc((size_t)EE * 4);
    float* tmp  = (float*)alloc((size_t)NN * HH * 4);
    float* x1   = (float*)alloc((size_t)NN * HH * 4);
    float* x2   = (float*)alloc((size_t)NN * HH * 4);
    float* x3   = (float*)alloc((size_t)NN * HH * 4);
    float* x4   = (float*)alloc((size_t)NN * 4);
    float* h4   = (float*)alloc((size_t)NN * 4);
    int*   sel  = (int*)  alloc((size_t)GG * KK * 4);
    float* xp   = (float*)alloc((size_t)GG * KK * DTOT * 4);
    float* c1o  = (float*)alloc((size_t)GG * 16 * POOLK * 4);
    float* c2o  = (float*)alloc((size_t)GG * 32 * 26 * 4);
    float* l1o  = (float*)alloc((size_t)GG * 128 * 4);
    float* x0   = x2;  // x0 dead before x2 is written (layer-2 scatter) -> safe alias

    const int B = 256;
    const int nB   = (NN + B - 1) / B;
    const int nhB  = (NN * HH) / B;                   // 50000, exact
    const int eB   = EE / B;                          // 12500, exact
    const int e16B = (EE * 16) / B;                   // 200000, exact

    // degrees + edge norms (shared across all layers)
    k_init_deg<<<nB, B, 0, stream>>>(dinv);
    k_count_deg<<<eB, B, 0, stream>>>(edst, dinv);
    k_rsqrt_deg<<<nB, B, 0, stream>>>(dinv);
    k_edge_norm<<<eB, B, 0, stream>>>(esrc, edst, dinv, nrm);

    // x0 = embedding
    k_embed<<<nhB, B, 0, stream>>>(z, z_table, x0);

    // three H=64 GCN layers (WMMA transform + norm-scatter + tanh)
    const float* xin[3]  = { x0, x1, x2 };
    float*       xout[3] = { x1, x2, x3 };
    const float* Ws[3]   = { W0, W1, W2 };
    const float* bs[3]   = { b0, b1, b2 };
    const int gemmB = (NN / 16) / 2;                  // 6250 blocks, 2 row-tiles each
    for (int l = 0; l < 3; ++l) {
        k_gemm64_wmma<<<gemmB, B, 0, stream>>>(xin[l], Ws[l], tmp, NN);
        k_self64<<<nhB, B, 0, stream>>>(tmp, dinv, xout[l]);
        k_scatter64<<<e16B, B, 0, stream>>>(esrc, edst, nrm, tmp, xout[l]);
        k_tanh_bias64<<<nhB, B, 0, stream>>>(xout[l], bs[l]);
    }

    // layer 4: H=1
    k_gemv64<<<nB, B, 0, stream>>>(x3, W3, h4);
    k_self1<<<nB, B, 0, stream>>>(h4, dinv, x4);
    k_scatter1<<<eB, B, 0, stream>>>(esrc, edst, nrm, h4, x4);
    k_tanh_bias1<<<nB, B, 0, stream>>>(x4, b3);

    // sort-pool + gather [G,K,193]
    k_sortpool<<<GG, B, 0, stream>>>(x4, sel);
    k_gather<<<GG * KK, 224, 0, stream>>>(sel, x1, x2, x3, x4, xp);

    // CNN + MLP head
    k_conv1pool<<<(GG * 16 * POOLK + B - 1) / B, B, 0, stream>>>(xp, c1w, c1b, c1o);
    k_conv2<<<(GG * 32 * 26 + B - 1) / B, B, 0, stream>>>(c1o, c2w, c2b, c2o);
    k_lin1<<<(GG * 128 + 127) / 128, 128, 0, stream>>>(c2o, l1w, l1b, l1o);
    k_lin2<<<(GG * 2 + B - 1) / B, B, 0, stream>>>(l1o, l2w, l2b, out);
}
